// StackedLinear_1803886265897
// MI455X (gfx1250) — matrix operand order repaired and checked by reading, not compiled, not run on hardware
//
#include <hip/hip_runtime.h>

typedef __attribute__((ext_vector_type(16))) __bf16 v16bf;
typedef __attribute__((ext_vector_type(8)))  __bf16 v8bf;
typedef __attribute__((ext_vector_type(8)))  float  v8f;
typedef __attribute__((ext_vector_type(4)))  float  v4f;

#define IN_F       2048
#define OUT_F      512
#define NEXP       8
#define TM         128
#define TN         128
#define TK         32
#define LDS_STRIDE 40   // TK + 8 bf16 pad -> 80B row stride, 16B aligned

// ---------------- bucketing kernels ----------------

__global__ void zero_counts_kernel(int* counts) {
  if (threadIdx.x < NEXP) counts[threadIdx.x] = 0;
}

__global__ void bucketize_kernel(const int* __restrict__ idx, int B,
                                 int* __restrict__ counts, int* __restrict__ bucket) {
  int i = blockIdx.x * blockDim.x + threadIdx.x;
  if (i < B) {
    int e = idx[i] & (NEXP - 1);
    int p = atomicAdd(&counts[e], 1);
    bucket[e * B + p] = i;
  }
}

// ---------------- split-precision helpers ----------------

__device__ __forceinline__ void cvt_split8(const float* f, v8bf& h, v8bf& l) {
#pragma unroll
  for (int i = 0; i < 8; ++i) {
    __bf16 hv = (__bf16)f[i];
    h[i] = hv;
    l[i] = (__bf16)(f[i] - (float)hv);
  }
}

// ---------------- grouped GEMM: out[b] = x[b] . W[e_b]^T + bias[e_b] ----------------

__global__ __launch_bounds__(256) void moe_wmma_gemm(
    const float* __restrict__ x, const float* __restrict__ W,
    const float* __restrict__ bias, const int* __restrict__ counts,
    const int* __restrict__ bucket, float* __restrict__ out, int B) {
  const int tilesPerExpert = (B + TM - 1) / TM;
  const int e        = blockIdx.x / tilesPerExpert;
  const int rowStart = (blockIdx.x % tilesPerExpert) * TM;
  const int cnt      = counts[e];
  if (rowStart >= cnt) return;                 // empty tile: uniform early exit
  const int nBase = blockIdx.y * TN;

  __shared__ __bf16 Ah[TM][LDS_STRIDE];
  __shared__ __bf16 Al[TM][LDS_STRIDE];
  __shared__ __bf16 Bh[TN][LDS_STRIDE];
  __shared__ __bf16 Bl[TN][LDS_STRIDE];
  __shared__ int    rowIds[TM];

  const int tid  = threadIdx.x;
  const int lane = tid & 31;                   // wave32
  const int wv   = tid >> 5;                   // 8 waves
  const int mw   = wv & 3;                     // 32-row strip of the 128-row tile
  const int nw   = wv >> 2;                    // 64-col strip of the 128-col tile
  const int l    = lane & 15;
  const int half = lane >> 4;                  // lane group 0-15 / 16-31

  if (tid < TM) {
    int p = rowStart + tid;
    rowIds[tid] = (p < cnt) ? bucket[e * B + p] : 0;   // clamp: junk rows masked on store
  }
  __syncthreads();

  // Per-lane LDS K offsets matching ISA 7.12.2 wave32 bf16 layouts.
  // A 16x32: lanes 0-15 hold K {0..7,16..23}; lanes 16-31 hold K {8..15,24..31}.
  const int kA0 = half ? 8 : 0;
  const int kA1 = kA0 + 16;
  // B 32x16: lanes 0-15 hold K {0..7,16..23}; lanes 16-31 hold K {8..15,24..31}.
  const int kB  = half ? 8 : 0;

  // Global->LDS staging map: 2 threads/row, 16 floats each, for both A (128 rows)
  // and B (128 rows of the expert's W slice).
  const int sr = tid >> 1;
  const int sk = (tid & 1) << 4;

  const float* xRow = x + (size_t)rowIds[sr] * IN_F + sk;
  const float* wRow = W + ((size_t)(e * OUT_F + nBase + sr)) * IN_F + sk;

  v8f acc[2][4] = {};

  // ---- software pipeline: global loads for stage k+1 overlap the WMMAs of stage k ----
  v4f ax[4], bw[4];
#pragma unroll
  for (int i = 0; i < 4; ++i) { ax[i] = *(const v4f*)(xRow + 4 * i); bw[i] = *(const v4f*)(wRow + 4 * i); }

  for (int k0 = 0; k0 < IN_F; k0 += TK) {
    // ---- convert held registers f32 -> (bf16 hi, bf16 lo), store to LDS ----
    {
      float fa[16], fw[16];
#pragma unroll
      for (int i = 0; i < 4; ++i) {
#pragma unroll
        for (int j = 0; j < 4; ++j) { fa[4 * i + j] = ax[i][j]; fw[4 * i + j] = bw[i][j]; }
      }
      v8bf h0, h1, l0, l1;
      cvt_split8(fa,     h0, l0);
      cvt_split8(fa + 8, h1, l1);
      *(v8bf*)&Ah[sr][sk]     = h0;
      *(v8bf*)&Ah[sr][sk + 8] = h1;
      *(v8bf*)&Al[sr][sk]     = l0;
      *(v8bf*)&Al[sr][sk + 8] = l1;
      cvt_split8(fw,     h0, l0);
      cvt_split8(fw + 8, h1, l1);
      *(v8bf*)&Bh[sr][sk]     = h0;
      *(v8bf*)&Bh[sr][sk + 8] = h1;
      *(v8bf*)&Bl[sr][sk]     = l0;
      *(v8bf*)&Bl[sr][sk + 8] = l1;
    }
    __syncthreads();

    // issue next stage's global loads now; latency hidden under the 24 WMMAs below
    const int kn = k0 + TK;
    if (kn < IN_F) {
#pragma unroll
      for (int i = 0; i < 4; ++i) {
        ax[i] = *(const v4f*)(xRow + kn + 4 * i);
        bw[i] = *(const v4f*)(wRow + kn + 4 * i);
      }
    }

    // ---- build fragments ----
    v16bf ahf[2], alf[2], bhf[4], blf[4];
#pragma unroll
    for (int i = 0; i < 2; ++i) {
      union { struct { v8bf a, b; } s; v16bf v; } u;
      const int r = mw * 32 + i * 16 + l;
      u.s.a = *(const v8bf*)&Ah[r][kA0]; u.s.b = *(const v8bf*)&Ah[r][kA1]; ahf[i] = u.v;
      u.s.a = *(const v8bf*)&Al[r][kA0]; u.s.b = *(const v8bf*)&Al[r][kA1]; alf[i] = u.v;
    }
#pragma unroll
    for (int i = 0; i < 4; ++i) {
      union { struct { v8bf a, b; } s; v16bf v; } u;
      const int c = nw * 64 + i * 16 + l;
      u.s.a = *(const v8bf*)&Bh[c][kB]; u.s.b = *(const v8bf*)&Bh[c][kB + 16]; bhf[i] = u.v;
      u.s.a = *(const v8bf*)&Bl[c][kB]; u.s.b = *(const v8bf*)&Bl[c][kB + 16]; blf[i] = u.v;
    }

    // ---- bf16x3 split-precision MMA: acc += al*bh + ah*bl + ah*bh ----
#pragma unroll
    for (int mi = 0; mi < 2; ++mi)
#pragma unroll
      for (int ni = 0; ni < 4; ++ni) {
        v8f t = __builtin_amdgcn_wmma_f32_16x16x32_bf16(
            false, alf[mi], false, bhf[ni], (short)0, acc[mi][ni], false, false);
        t     = __builtin_amdgcn_wmma_f32_16x16x32_bf16(
            false, ahf[mi], false, blf[ni], (short)0, t,           false, false);
        acc[mi][ni] = __builtin_amdgcn_wmma_f32_16x16x32_bf16(
            false, ahf[mi], false, bhf[ni], (short)0, t,           false, false);
      }
    __syncthreads();
  }

  // ---- epilogue: add bias, scatter rows back to original sample ids ----
  const bool fullTile = (rowStart + TM) <= cnt;
#pragma unroll
  for (int mi = 0; mi < 2; ++mi) {
#pragma unroll
    for (int ni = 0; ni < 4; ++ni) {
      const int col = nBase + nw * 64 + ni * 16 + l;
      const float bs = bias[e * OUT_F + col];
      if (fullTile) {
#pragma unroll
        for (int j = 0; j < 8; ++j) {
          const int rit = mw * 32 + mi * 16 + j + (half << 3);   // C/D layout: VGPR j, halves
          __builtin_nontemporal_store(acc[mi][ni][j] + bs,
                                      &out[(size_t)rowIds[rit] * OUT_F + col]);
        }
      } else {
#pragma unroll
        for (int j = 0; j < 8; ++j) {
          const int rit = mw * 32 + mi * 16 + j + (half << 3);
          if (rowStart + rit < cnt) {
            __builtin_nontemporal_store(acc[mi][ni][j] + bs,
                                        &out[(size_t)rowIds[rit] * OUT_F + col]);
          }
        }
      }
    }
  }
}

// ---------------- launcher ----------------

extern "C" void kernel_launch(void* const* d_in, const int* in_sizes, int n_in,
                              void* d_out, int out_size, void* d_ws, size_t ws_size,
                              hipStream_t stream) {
  const float* x    = (const float*)d_in[0];
  const int*   idx  = (const int*)d_in[1];
  const float* W    = (const float*)d_in[2];
  const float* bias = (const float*)d_in[3];
  float* out = (float*)d_out;
  const int B = in_sizes[1];

  int* counts = (int*)d_ws;       // 8 ints
  int* bucket = counts + 16;      // NEXP * B ints

  zero_counts_kernel<<<1, 32, 0, stream>>>(counts);
  bucketize_kernel<<<(B + 255) / 256, 256, 0, stream>>>(idx, B, counts, bucket);

  const int tilesPerExpert = (B + TM - 1) / TM;
  dim3 grid(NEXP * tilesPerExpert, OUT_F / TN);
  moe_wmma_gemm<<<grid, 256, 0, stream>>>(x, W, bias, counts, bucket, out, B);
}
